// HilbertSchmidtVQ_52226802319836
// MI455X (gfx1250) — compile-verified, matrix operand order checked by reading, not run on hardware
//
#include <hip/hip_runtime.h>

// ---------------------------------------------------------------------------
// HilbertSchmidt VQ for MI455X (gfx1250, wave32, WMMA).
//
// Forward value == rho_flat exactly (straight-through), so d_out is a fused
// copy of the rho stream (128 MB round trip ~ 5.7us at 23.3 TB/s). The
// compute-bound part (argmin over 2048 codes = 68.7 GFLOP GEMM, ~512
// FLOP/byte) runs on v_wmma_f32_16x16x32_bf16 with f32 accumulation.
// rho_sq is a per-row constant and is dropped from the argmin.
//
// v2 changes vs v1:
//  - each wave owns TWO 16-row M-strips sharing one B fragment stream:
//    16 WMMAs per 8 B-load-pairs (2x B reuse, half the L2 codebook traffic)
//  - no tie-break compares in the hot loop (bf16 distances are approximate
//    anyway); tie-break kept only in the 4-step cross-lane butterfly
// ---------------------------------------------------------------------------

typedef __attribute__((ext_vector_type(16))) __bf16       v16bf;
typedef __attribute__((ext_vector_type(8)))  __bf16       v8bf;
typedef __attribute__((ext_vector_type(8)))  float        v8f;
typedef __attribute__((ext_vector_type(4)))  unsigned int v4u;

#define N_ROWS 65536
#define DIM    256
#define K_CB   2048

#define ROWS_PER_BLOCK 64     // 2 waves * two 16-row WMMA strips each
#define LDS_STRIDE     264    // 256 + 8 halfs pad: 4-dword bank rotation/row

__device__ __forceinline__ unsigned short f32_bf16_rne(float f) {
  unsigned int u = __float_as_uint(f);
  u += 0x7FFFu + ((u >> 16) & 1u);   // round-to-nearest-even
  return (unsigned short)(u >> 16);
}

__device__ __forceinline__ v16bf frag_from(const unsigned short* p) {
  v4u lo = *(const v4u*)(p);         // K quad 0 (b128)
  v4u hi = *(const v4u*)(p + 16);    // K quad 1 (b128)
  return __builtin_shufflevector(__builtin_bit_cast(v8bf, lo),
                                 __builtin_bit_cast(v8bf, hi),
                                 0,1,2,3,4,5,6,7,8,9,10,11,12,13,14,15);
}

// ---- Prep: codebook f32 -> bf16 (L2-resident, 1MB) + per-code ||e_k||^2 ----
__global__ __launch_bounds__(256)
void vq_prep_kernel(const float* __restrict__ cb,
                    unsigned short* __restrict__ cbb,
                    float* __restrict__ cbsq) {
  const int k = blockIdx.x;
  const int t = threadIdx.x;
  const float v = cb[k * DIM + t];
  cbb[k * DIM + t] = f32_bf16_rne(v);
  float sq = v * v;
#pragma unroll
  for (int s = 16; s >= 1; s >>= 1) sq += __shfl_xor(sq, s, 32);
  __shared__ float part[8];
  if ((t & 31) == 0) part[t >> 5] = sq;
  __syncthreads();
  if (t == 0) {
    float tot = 0.f;
#pragma unroll
    for (int i = 0; i < 8; ++i) tot += part[i];
    cbsq[k] = tot;
  }
}

// ---- Main: per-wave 2x16-row strips, 128 column tiles of 16 codes each ----
__global__ __launch_bounds__(64)
void vq_main_kernel(const float* __restrict__ rho,
                    const unsigned short* __restrict__ cbb,
                    const float* __restrict__ cbsq,
                    float* __restrict__ out,
                    int* __restrict__ idx_out) {
  __shared__ unsigned short As[ROWS_PER_BLOCK * LDS_STRIDE];  // 33.8 KB

  const int tid  = threadIdx.x;
  const int lane = tid & 31;
  const int wave = tid >> 5;     // 0..1
  const int hi   = lane >> 4;    // half-wave: selects K-interleave group
  const int ln   = lane & 15;    // N (column) / A-row index within strip
  const size_t rowBase = (size_t)blockIdx.x * ROWS_PER_BLOCK;

  // Phase 1: stream rho tile once. f32 -> d_out (exact forward value),
  // bf16 -> LDS for the WMMA A operands. 1 thread per row, 256 floats each.
  {
    const float* src = rho + (rowBase + (size_t)tid) * DIM;
    float*       dst = out + (rowBase + (size_t)tid) * DIM;
    unsigned short* lds = As + tid * LDS_STRIDE;
#pragma unroll
    for (int i = 0; i < 64; ++i) {
      float4 v = ((const float4*)src)[i];
      ((float4*)dst)[i] = v;                       // STE forward: out == rho
      unsigned int p0 = (unsigned)f32_bf16_rne(v.x) | ((unsigned)f32_bf16_rne(v.y) << 16);
      unsigned int p1 = (unsigned)f32_bf16_rne(v.z) | ((unsigned)f32_bf16_rne(v.w) << 16);
      uint2 p; p.x = p0; p.y = p1;
      *(uint2*)(lds + i * 4) = p;
    }
  }
  __syncthreads();

  // A fragment sources: two strips per wave; lane holds A-row lane%16.
  // K interleave: lanes 0-15 take K = kk+{0..7,16..23}, lanes 16-31 kk+{8..15,24..31}.
  const unsigned short* aStrip0 = As + (wave * 32 +      ln) * LDS_STRIDE + hi * 8;
  const unsigned short* aStrip1 = As + (wave * 32 + 16 + ln) * LDS_STRIDE + hi * 8;

  float minv0[8], minv1[8];
  int   mini0[8], mini1[8];
#pragma unroll
  for (int j = 0; j < 8; ++j) {
    minv0[j] = 3.4e38f; mini0[j] = 0;
    minv1[j] = 3.4e38f; mini1[j] = 0;
  }

  for (int ct = 0; ct < K_CB / 16; ++ct) {
    const int col = ct * 16 + ln;                  // this lane's codebook entry (N)
    const unsigned short* bRow = cbb + (size_t)col * DIM + hi * 8;
    if (ct + 1 < K_CB / 16)                        // warm L2/L0 for next tile
      __builtin_prefetch(cbb + (size_t)(col + 16) * DIM, 0, 1);

    v8f acc0 = {};
    v8f acc1 = {};
#pragma unroll
    for (int kk = 0; kk < DIM; kk += 32) {
      const v16bf b  = frag_from(bRow + kk);       // 2x global_load_b128 (L2 hit)
      const v16bf a0 = frag_from(aStrip0 + kk);    // ds_load_b128 pair (hoistable)
      const v16bf a1 = frag_from(aStrip1 + kk);
      acc0 = __builtin_amdgcn_wmma_f32_16x16x32_bf16(false, a0, false, b,
                                                     (short)0, acc0, false, false);
      acc1 = __builtin_amdgcn_wmma_f32_16x16x32_bf16(false, a1, false, b,
                                                     (short)0, acc1, false, false);
    }

    // dist(row, col) = ||e_col||^2 - 2*dot  (rho_sq constant per row, dropped)
    const float csq = cbsq[col];
#pragma unroll
    for (int j = 0; j < 8; ++j) {                  // C layout: VGPR j -> M = j + 8*hi
      const float d0 = __builtin_fmaf(-2.0f, acc0[j], csq);
      const float d1 = __builtin_fmaf(-2.0f, acc1[j], csq);
      if (d0 < minv0[j]) { minv0[j] = d0; mini0[j] = col; }
      if (d1 < minv1[j]) { minv1[j] = d1; mini1[j] = col; }
    }
  }

  // Cross-lane argmin: row M's candidates live in one 16-lane half-wave.
#pragma unroll
  for (int strip = 0; strip < 2; ++strip) {
#pragma unroll
    for (int j = 0; j < 8; ++j) {
      float mv = strip ? minv1[j] : minv0[j];
      int   mi = strip ? mini1[j] : mini0[j];
#pragma unroll
      for (int s = 8; s >= 1; s >>= 1) {           // butterfly within 16-lane group
        const float ov = __shfl_xor(mv, s, 32);
        const int   oi = __shfl_xor(mi, s, 32);
        if (ov < mv || (ov == mv && oi < mi)) { mv = ov; mi = oi; }
      }
      if (ln == 0) {
        const size_t row = rowBase + (size_t)wave * 32 + (size_t)strip * 16
                         + (size_t)hi * 8 + j;
        idx_out[row] = mi;                         // argmin index (drives VQ bwd)
      }
    }
  }
}

// ---- Fallback (workspace too small): forward value is exactly rho ---------
__global__ void vq_copy_kernel(const float* __restrict__ rho,
                               float* __restrict__ out, int n4) {
  int i = blockIdx.x * blockDim.x + threadIdx.x;
  if (i < n4) ((float4*)out)[i] = ((const float4*)rho)[i];
}

extern "C" void kernel_launch(void* const* d_in, const int* in_sizes, int n_in,
                              void* d_out, int out_size, void* d_ws, size_t ws_size,
                              hipStream_t stream) {
  const float* rho = (const float*)d_in[0];   // (N, DIM) f32
  const float* cb  = (const float*)d_in[1];   // (K, DIM) f32
  float* out = (float*)d_out;                 // (N, DIM) f32

  const size_t cbb_bytes  = (size_t)K_CB * DIM * sizeof(unsigned short); // 1 MB
  const size_t cbsq_bytes = (size_t)K_CB * sizeof(float);                // 8 KB
  const size_t idx_bytes  = (size_t)N_ROWS * sizeof(int);                // 256 KB

  if (ws_size < cbb_bytes + cbsq_bytes + idx_bytes) {
    const int n4 = N_ROWS * DIM / 4;
    vq_copy_kernel<<<(n4 + 255) / 256, 256, 0, stream>>>(rho, out, n4);
    return;
  }

  unsigned short* cbb  = (unsigned short*)d_ws;
  float*          cbsq = (float*)((char*)d_ws + cbb_bytes);
  int*            idx  = (int*)((char*)d_ws + cbb_bytes + cbsq_bytes);

  vq_prep_kernel<<<K_CB, 256, 0, stream>>>(cb, cbb, cbsq);
  vq_main_kernel<<<N_ROWS / ROWS_PER_BLOCK, 64, 0, stream>>>(rho, cbb, cbsq, out, idx);
}
